// Head_73899207295023
// MI455X (gfx1250) — matrix operand order repaired and checked by reading, not compiled, not run on hardware
//
#include <hip/hip_runtime.h>
#include <hip/hip_bf16.h>

// ---------------------------------------------------------------------------
// Fused causal single-head attention with RoPE for MI455X (gfx1250, wave32).
// Phase 1: QKV projection GEMM via v_wmma_f32_16x16x32_bf16 + in-register RoPE,
//          writes bf16 Q/K/V to workspace (6 MB).
// Phase 2: flash attention (online softmax), S and P·V via bf16 WMMA.
// ---------------------------------------------------------------------------

#define B_ 8
#define T_ 2048
#define C_ 1024
#define H_ 64

typedef __attribute__((ext_vector_type(16))) __bf16    v16bf;
typedef __attribute__((ext_vector_type(8)))  float     v8f;
typedef __attribute__((ext_vector_type(8)))  unsigned  v8u;

static __device__ __forceinline__ unsigned short f2bf(float f) {
    unsigned u = __builtin_bit_cast(unsigned, f);
    unsigned r = u + 0x7FFFu + ((u >> 16) & 1u);   // round-to-nearest-even
    return (unsigned short)(r >> 16);
}
static __device__ __forceinline__ unsigned pack2bf(float lo, float hi) {
    return (unsigned)f2bf(lo) | ((unsigned)f2bf(hi) << 16);
}
static __device__ __forceinline__ v8f vzero8() {
    v8f z;
#pragma unroll
    for (int e = 0; e < 8; ++e) z[e] = 0.0f;
    return z;
}

// ---------------------------------------------------------------------------
// Phase 1: Q/K/V projection + RoPE.
// Grid: 128 blocks x 256 threads (8 waves). Each wave owns one 16-row tile of
// (b, t); accumulates 16x64 Q, K, V tiles over C=1024 in K-chunks of 32.
// W chunks staged in LDS pre-packed as bf16 pairs in B-fragment order.
// ---------------------------------------------------------------------------
__global__ __launch_bounds__(256) void qkv_rope_kernel(
    const float* __restrict__ x,
    const float* __restrict__ Wq,
    const float* __restrict__ Wk,
    const float* __restrict__ Wv,
    unsigned short* __restrict__ Qb,
    unsigned short* __restrict__ Kb,
    unsigned short* __restrict__ Vb)
{
    // sW[mat][n*17 + p]: packed {W[2p,n], W[2p+1,n]} bf16 pair, 17-stride pad
    __shared__ unsigned sW[3 * 1088];

    const int tid   = threadIdx.x;
    const int lane  = tid & 31;
    const int wave  = tid >> 5;
    const int tile  = blockIdx.x * 8 + wave;        // 0..1023
    const int bIdx  = tile >> 7;                    // / (T/16)
    const int t0    = (tile & 127) << 4;
    const int M     = lane & 15;                    // A-frag row, also N for B
    const int khalf = lane >> 4;

    const float* Ws[3] = { Wq, Wk, Wv };

    v8f acc[12];                                    // [mat 0..2][ntile 0..3]
#pragma unroll
    for (int i = 0; i < 12; ++i) acc[i] = vzero8();

    for (int kc = 0; kc < 32; ++kc) {
        const int c0 = kc * 32;

        // --- cooperative stage of 3 x (32 x 64) W chunk, bf16-packed -------
#pragma unroll
        for (int m = 0; m < 3; ++m) {
            const float* W = Ws[m];
#pragma unroll
            for (int i = 0; i < 4; ++i) {
                int pairIdx = i * 256 + tid;        // 1024 pairs per matrix
                int n = pairIdx & 63;
                int p = pairIdx >> 6;               // K-pair index 0..15
                float w0 = W[(size_t)(c0 + 2 * p)     * H_ + n];
                float w1 = W[(size_t)(c0 + 2 * p + 1) * H_ + n];
                sW[m * 1088 + n * 17 + p] = pack2bf(w0, w1);
            }
        }
        __syncthreads();

        // --- A fragment (16x32 bf16) from x, fp32 -> bf16 in-register ------
        v8u af;
#pragma unroll
        for (int j = 0; j < 8; ++j) {
            int K = 2 * j + 16 * khalf;
            const float2 xv = *(const float2*)&x[(size_t)(bIdx * T_ + t0 + M) * C_ + c0 + K];
            af[j] = pack2bf(xv.x, xv.y);
        }
        v16bf a = __builtin_bit_cast(v16bf, af);

        // --- 12 WMMAs: 3 matrices x 4 N-tiles ------------------------------
#pragma unroll
        for (int m = 0; m < 3; ++m) {
#pragma unroll
            for (int nt = 0; nt < 4; ++nt) {
                v8u bu;
#pragma unroll
                for (int j = 0; j < 8; ++j) {
                    int p = 8 * khalf + j;          // K-pair for this half
                    int n = nt * 16 + M;            // output column
                    bu[j] = sW[m * 1088 + n * 17 + p];
                }
                v16bf bb = __builtin_bit_cast(v16bf, bu);
                acc[m * 4 + nt] = __builtin_amdgcn_wmma_f32_16x16x32_bf16(
                    false, a, false, bb, (short)0, acc[m * 4 + nt], false, false);
            }
        }
        __syncthreads();
    }

    // --- RoPE on Q, K accumulators; store bf16 Q/K/V ----------------------
    const int N = lane & 15;
#pragma unroll
    for (int nt = 0; nt < 4; ++nt) {
        int   ng    = nt * 16 + N;                              // h index 0..63
        float invf  = __expf(-(float)(ng >> 1) * (9.210340371976184f / 32.0f));
        bool  evenN = (ng & 1) == 0;
#pragma unroll
        for (int r = 0; r < 8; ++r) {
            int    tRow = t0 + r + 8 * khalf;                   // row for VGPR r
            float  th   = (float)tRow * invf;
            float  ct   = __cosf(th), st = __sinf(th);
            size_t oidx = (size_t)(bIdx * T_ + tRow) * H_ + ng;
            {   // Q with RoPE (partner element lives in lane^1)
                float v0  = acc[0 * 4 + nt][r];
                float pr  = __shfl_xor(v0, 1, 32);
                float rot = evenN ? -pr : pr;
                Qb[oidx]  = f2bf(v0 * ct + rot * st);
            }
            {   // K with RoPE
                float v0  = acc[1 * 4 + nt][r];
                float pr  = __shfl_xor(v0, 1, 32);
                float rot = evenN ? -pr : pr;
                Kb[oidx]  = f2bf(v0 * ct + rot * st);
            }
            Vb[oidx] = f2bf(acc[2 * 4 + nt][r]);                // V unrotated
        }
    }
}

// ---------------------------------------------------------------------------
// Phase 2: flash attention. Grid: 256 blocks x 128 threads (4 waves).
// Each wave owns one 16-row query tile; loops over 16-key tiles up to the
// causal diagonal with online softmax. S and P·V via bf16 WMMA.
// ---------------------------------------------------------------------------
__global__ __launch_bounds__(128) void attn_kernel(
    const unsigned short* __restrict__ Qb,
    const unsigned short* __restrict__ Kb,
    const unsigned short* __restrict__ Vb,
    float* __restrict__ out)
{
    __shared__ float sP[4][16 * 17];                // per-wave private slices

    const int tid   = threadIdx.x;
    const int lane  = tid & 31;
    const int w     = tid >> 5;
    const int tile  = blockIdx.x * 4 + w;           // 0..1023
    const int bIdx  = tile >> 7;
    const int t0    = (tile & 127) << 4;
    const int N     = lane & 15;
    const int khalf = lane >> 4;

    const unsigned* Qw = (const unsigned*)Qb;       // bf16 pairs
    const unsigned* Kw = (const unsigned*)Kb;

    // --- Q as two A fragments (16 x 32 bf16 each), loaded once -------------
    v8u qa0u, qa1u;
#pragma unroll
    for (int j = 0; j < 8; ++j) {
        int    K  = 2 * j + 16 * khalf;
        size_t e0 = (size_t)(bIdx * T_ + t0 + N) * H_ + K;      // N = A row M
        qa0u[j] = Qw[e0 >> 1];
        qa1u[j] = Qw[(e0 + 32) >> 1];
    }
    v16bf qa0 = __builtin_bit_cast(v16bf, qa0u);
    v16bf qa1 = __builtin_bit_cast(v16bf, qa1u);

    v8f o[4];
#pragma unroll
    for (int nt = 0; nt < 4; ++nt) o[nt] = vzero8();
    float m_[8], l_[8];
#pragma unroll
    for (int r = 0; r < 8; ++r) { m_[r] = -3.0e38f; l_[r] = 0.0f; }

    const int jmax = t0 >> 4;
    for (int j = 0; j <= jmax; ++j) {
        const int s0 = j << 4;

        // --- K tile as B fragments (h-major pairs are contiguous) ----------
        v8u kb0u, kb1u;
#pragma unroll
        for (int jj = 0; jj < 8; ++jj) {
            int    h0 = 16 * khalf + 2 * jj;
            size_t e0 = (size_t)(bIdx * T_ + s0 + N) * H_ + h0; // N = key idx
            kb0u[jj] = Kw[e0 >> 1];
            kb1u[jj] = Kw[(e0 + 32) >> 1];
        }
        if (j < jmax) {  // prefetch next key tile row into cache
            __builtin_prefetch(&Kb[(size_t)(bIdx * T_ + s0 + 16 + N) * H_], 0, 0);
            __builtin_prefetch(&Vb[(size_t)(bIdx * T_ + s0 + 16 + N) * H_], 0, 0);
        }
        v16bf kb0 = __builtin_bit_cast(v16bf, kb0u);
        v16bf kb1 = __builtin_bit_cast(v16bf, kb1u);

        v8f S = vzero8();
        S = __builtin_amdgcn_wmma_f32_16x16x32_bf16(false, qa0, false, kb0, (short)0, S, false, false);
        S = __builtin_amdgcn_wmma_f32_16x16x32_bf16(false, qa1, false, kb1, (short)0, S, false, false);

        // --- online softmax: row M = r + 8*khalf, cols across 16 lanes -----
        float p_[8], alpha[8];
#pragma unroll
        for (int r = 0; r < 8; ++r) {
            float s = S[r] * 0.125f;                            // 1/sqrt(64)
            if (j == jmax && (s0 + N) > (t0 + r + 8 * khalf)) s = -3.0e38f;
            float rm = s;
            rm = fmaxf(rm, __shfl_xor(rm, 1, 32));
            rm = fmaxf(rm, __shfl_xor(rm, 2, 32));
            rm = fmaxf(rm, __shfl_xor(rm, 4, 32));
            rm = fmaxf(rm, __shfl_xor(rm, 8, 32));
            float mn = fmaxf(m_[r], rm);
            float al = __expf(m_[r] - mn);
            float p  = __expf(s - mn);
            float rs = p;
            rs += __shfl_xor(rs, 1, 32);
            rs += __shfl_xor(rs, 2, 32);
            rs += __shfl_xor(rs, 4, 32);
            rs += __shfl_xor(rs, 8, 32);
            l_[r]    = l_[r] * al + rs;
            m_[r]    = mn;
            alpha[r] = al;
            p_[r]    = p;
        }
#pragma unroll
        for (int nt = 0; nt < 4; ++nt)
#pragma unroll
            for (int r = 0; r < 8; ++r) o[nt][r] *= alpha[r];

        // --- C-layout -> A-fragment transpose of P via padded LDS ----------
#pragma unroll
        for (int r = 0; r < 8; ++r)
            sP[w][(r + 8 * khalf) * 17 + N] = p_[r];
        v8u pau;
#pragma unroll
        for (int jj = 0; jj < 4; ++jj) {
            int K0  = 8 * khalf + 2 * jj;
            pau[jj] = pack2bf(sP[w][N * 17 + K0], sP[w][N * 17 + K0 + 1]);
        }
#pragma unroll
        for (int jj = 4; jj < 8; ++jj) pau[jj] = 0u;            // pad K 16..31
        v16bf pa = __builtin_bit_cast(v16bf, pau);

        // --- P·V: 4 output tiles of 16 columns -----------------------------
#pragma unroll
        for (int nt = 0; nt < 4; ++nt) {
            v8u vbu;
#pragma unroll
            for (int jj = 0; jj < 8; ++jj) {
                unsigned val = 0u;
                if (jj < 4) {                                   // only K 0..15 live
                    int    K0 = 8 * khalf + 2 * jj;
                    size_t e0 = (size_t)(bIdx * T_ + s0 + K0) * H_ + nt * 16 + N;
                    unsigned short lo = Vb[e0];
                    unsigned short hi = Vb[e0 + H_];
                    val = (unsigned)lo | ((unsigned)hi << 16);
                }
                vbu[jj] = val;
            }
            v16bf vb = __builtin_bit_cast(v16bf, vbu);
            o[nt] = __builtin_amdgcn_wmma_f32_16x16x32_bf16(
                false, pa, false, vb, (short)0, o[nt], false, false);
        }
    }

    // --- normalize and store fp32 output ----------------------------------
#pragma unroll
    for (int r = 0; r < 8; ++r) {
        float inv  = 1.0f / l_[r];
        int   tRow = t0 + r + 8 * khalf;
#pragma unroll
        for (int nt = 0; nt < 4; ++nt)
            out[(size_t)(bIdx * T_ + tRow) * H_ + nt * 16 + N] = o[nt][r] * inv;
    }
}

// ---------------------------------------------------------------------------
extern "C" void kernel_launch(void* const* d_in, const int* in_sizes, int n_in,
                              void* d_out, int out_size, void* d_ws, size_t ws_size,
                              hipStream_t stream)
{
    (void)in_sizes; (void)n_in; (void)out_size;

    const float* x  = (const float*)d_in[0];
    const float* Wq = (const float*)d_in[1];
    const float* Wk = (const float*)d_in[2];
    const float* Wv = (const float*)d_in[3];

    const size_t elems = (size_t)B_ * T_ * H_;      // per Q/K/V buffer
    if (ws_size < 3 * elems * sizeof(unsigned short)) return;

    unsigned short* Qb = (unsigned short*)d_ws;
    unsigned short* Kb = Qb + elems;
    unsigned short* Vb = Kb + elems;

    // Phase 1: 1024 tiles of 16 rows, 8 waves per block of 256 threads.
    qkv_rope_kernel<<<128, 256, 0, stream>>>(x, Wq, Wk, Wv, Qb, Kb, Vb);
    // Phase 2: 1024 query tiles, 4 waves per block of 128 threads.
    attn_kernel<<<256, 128, 0, stream>>>(Qb, Kb, Vb, (float*)d_out);
}
